// DotProdSelfAttention_62723702391430
// MI455X (gfx1250) — matrix operand order repaired and checked by reading, not compiled, not run on hardware
//
#include <hip/hip_runtime.h>

typedef __bf16 bf16;
typedef __attribute__((ext_vector_type(16))) __bf16 v16bf;
typedef __attribute__((ext_vector_type(8)))  __bf16 v8bf;
typedef __attribute__((ext_vector_type(4)))  __bf16 v4bf;
typedef __attribute__((ext_vector_type(8)))  float  v8f;
typedef __attribute__((ext_vector_type(4)))  unsigned int u32x4;
typedef __attribute__((ext_vector_type(8)))  int i32x8;
typedef __attribute__((ext_vector_type(4)))  int i32x4;

union Frag { v16bf f; v8bf h[2]; };

static constexpr int Bn = 4, S = 2048, D = 768, U = 768;
static constexpr int R  = Bn * S;        // 8192 flat rows
static constexpr int DT = D / 32;        // 24 contraction steps (k=32 per WMMA)
static constexpr int UT = U / 16;        // 48 output column tiles
static constexpr int KC = S / 32;        // 64 key chunks per batch
static constexpr int VCHUNK = UT * 32 * 16;   // 24576 bf16 elems = 48KB per key chunk

// Workspace layout (bytes). Total = 53,870,592 (~51.4 MB, fits in 192MB L2).
static constexpr size_t XB_OFF = 0;                       // x bf16 row-major     12,582,912
static constexpr size_t WT_OFF = 12582912;                // W tiled B-fragments   3,538,944
static constexpr size_t QB_OFF = 16121856;                // Q bf16 row-major     12,582,912
static constexpr size_t KB_OFF = 28704768;                // K bf16 row-major     12,582,912
static constexpr size_t VT_OFF = 41287680;                // V tiled B-fragments  12,582,912

__device__ __forceinline__ v8f wmma_bf16(v16bf a, v16bf b, v8f c) {
  // D = A(16x32 bf16) * B(32x16 bf16) + C(16x16 f32)
  return __builtin_amdgcn_wmma_f32_16x16x32_bf16(false, a, false, b, (short)0, c,
                                                 false, false);
}

// ---------------------------------------------------------------------------
// TDM: DMA one contiguous 48KB V chunk (24576 bf16) from global to LDS.
// D# per CDNA5 ISA:
//   group0: [1:0]=count=1 | [63:32]=lds_addr | [120:64]=global_addr | [127:126]=type 2
//   group1: data_size=2B(code 1), tensor_dim0=tile_dim0=24576, tensor_dim1=tile_dim1=1,
//           tile_dim2=0, dim0_stride=dim1_stride=24576
//   remaining groups: zero (<=2D tensor)
// ---------------------------------------------------------------------------
__device__ __forceinline__ void tdm_load_vchunk(const bf16* gsrc, unsigned int lds_addr) {
  unsigned long long ga = (unsigned long long)gsrc;
  u32x4 g0 = { 1u,                                   // count=1, user mode
               lds_addr,                             // lds_addr[31:0]
               (unsigned int)ga,                     // global_addr[31:0]
               (unsigned int)((ga >> 32) & 0x01FFFFFFu) | (2u << 30) }; // addr[56:32]|type=2
  i32x8 g1 = { (int)(1u << 16),                      // wg_mask=0, data_size=1 (2 bytes)
               (int)(0x6000u << 16),                 // tensor_dim0[15:0]=24576 in [63:48]
               (int)(1u << 16),                      // tensor_dim0[31:16]=0, tensor_dim1=1
               (int)(0x6000u << 16),                 // tensor_dim1 hi=0, tile_dim0=24576
               1,                                    // tile_dim1=1, tile_dim2=0
               24576,                                // tensor_dim0_stride[31:0]
               (int)(0x6000u << 16),                 // stride0 hi=0, stride1[15:0]=24576
               0 };                                  // stride1[47:16]=0
  i32x4 z4 = { 0, 0, 0, 0 };
  i32x8 z8 = { 0, 0, 0, 0, 0, 0, 0, 0 };
  // clang-23 / therock toolchain: 6-arg form (u32x4, i32x8, i32x4, i32x4, i32x8, cpol)
  __builtin_amdgcn_tensor_load_to_lds(g0, g1, z4, z4, z8, 0);
}

// ---------------------------------------------------------------------------
// Kernel 0a: x f32 -> bf16 row-major
// ---------------------------------------------------------------------------
__global__ void cvt_x_kernel(const float* __restrict__ x, bf16* __restrict__ xb, int n4) {
  int t = blockIdx.x * blockDim.x + threadIdx.x;
  if (t >= n4) return;
  const float4 v = reinterpret_cast<const float4*>(x)[t];
  v4bf o = { (bf16)v.x, (bf16)v.y, (bf16)v.z, (bf16)v.w };
  reinterpret_cast<v4bf*>(xb)[t] = o;
}

// ---------------------------------------------------------------------------
// Kernel 0b: Wq/Wk/Wv f32 -> bf16 tiled into WMMA B-fragment layout.
// Wt[((z*DT + dt)*UT + ut)*32 + L][i] = W_z[dt*32 + (L/16)*16 + i][ut*16 + L%16]
// ---------------------------------------------------------------------------
__global__ void cvt_w_kernel(const float* __restrict__ Wq, const float* __restrict__ Wk,
                             const float* __restrict__ Wv, bf16* __restrict__ Wt) {
  int gid  = blockIdx.x * blockDim.x + threadIdx.x;   // tile*32 + L
  int L    = gid & 31;
  int tile = gid >> 5;                                // ((z*DT + dt)*UT + ut)
  if (tile >= 3 * DT * UT) return;
  int ut = tile % UT;
  int dt = (tile / UT) % DT;
  int z  = tile / (UT * DT);
  const float* W = (z == 0) ? Wq : (z == 1) ? Wk : Wv;
  int drow = dt * 32 + (L >> 4) * 16;
  int col  = ut * 16 + (L & 15);
  bf16* dst = Wt + (size_t)tile * 32 * 16 + (size_t)L * 16;
#pragma unroll
  for (int i = 0; i < 16; ++i)
    dst[i] = (bf16)W[(size_t)(drow + i) * U + col];
}

// ---------------------------------------------------------------------------
// Kernel 1: QKV projection GEMM.  grid (R/16, UT/4, 3), block 128 (4 waves).
// ---------------------------------------------------------------------------
__global__ void __launch_bounds__(128, 1)
qkv_gemm_kernel(const bf16* __restrict__ xb, const bf16* __restrict__ Wt,
                bf16* __restrict__ Qb, bf16* __restrict__ Kb, bf16* __restrict__ Vt) {
  const int lane = threadIdx.x & 31;
  const int wave = threadIdx.x >> 5;
  const int h = lane >> 4, n = lane & 15;
  const int rt = blockIdx.x;                 // row tile [0,512)
  const int ut = blockIdx.y * 4 + wave;      // u tile   [0,48)
  const int z  = blockIdx.z;                 // 0=Q 1=K 2=V

  const bf16* arow = xb + (size_t)(rt * 16 + n) * D;
  const bf16* bfr  = Wt + ((size_t)((z * DT) * UT + ut) * 32 + lane) * 16;

  v8f c = { 0.f, 0.f, 0.f, 0.f, 0.f, 0.f, 0.f, 0.f };
#pragma unroll
  for (int dt = 0; dt < DT; ++dt) {
    Frag a, b;
    a.h[0] = *reinterpret_cast<const v8bf*>(arow + dt * 32 + h * 16);
    a.h[1] = *reinterpret_cast<const v8bf*>(arow + dt * 32 + h * 16 + 8);
    b.h[0] = *reinterpret_cast<const v8bf*>(bfr + (size_t)dt * UT * 32 * 16);
    b.h[1] = *reinterpret_cast<const v8bf*>(bfr + (size_t)dt * UT * 32 * 16 + 8);
    c = wmma_bf16(a.f, b.f, c);
  }

  const float qscale = 1.4426950408889634f / 27.712812921102035f; // log2e/sqrt(U)

  if (z == 2) {
#pragma unroll
    for (int r = 0; r < 8; ++r) {
      int row = rt * 16 + r + 8 * h;
      int bb  = row >> 11;
      int s   = row & (S - 1);
      int kt  = s >> 5;
      int rr  = s & 31;
      size_t idx = (((size_t)(bb * KC + kt) * UT + ut) * 32 + (n + 16 * ((rr >> 3) & 1))) * 16
                   + ((rr & 7) + 8 * (rr >> 4));
      Vt[idx] = (bf16)c[r];
    }
  } else {
    bf16* dst = (z == 0) ? Qb : Kb;
    const float sc = (z == 0) ? qscale : 1.0f;
#pragma unroll
    for (int r = 0; r < 8; ++r) {
      int row = rt * 16 + r + 8 * h;
      dst[(size_t)row * U + ut * 16 + n] = (bf16)(c[r] * sc);
    }
  }
}

// ---------------------------------------------------------------------------
// Kernel 2: streaming (flash) attention. grid 256, block 64 (2 waves).
// V chunks are DMA'd into LDS by the Tensor Data Mover, double-buffered one
// chunk ahead; both waves of the block consume the shared copy via ds_load_b128.
// Dynamic LDS: 2 x 48KB V buffers + 2 x 1KB P-transpose staging = 100,352 B.
// ---------------------------------------------------------------------------
__global__ void __launch_bounds__(64, 1)
attn_kernel(const bf16* __restrict__ Qb, const bf16* __restrict__ Kb,
            const bf16* __restrict__ Vt, float* __restrict__ out) {
  extern __shared__ char smem[];
  bf16* vbuf = (bf16*)smem;                         // [2][VCHUNK]
  const int lane = threadIdx.x & 31;
  const int wave = threadIdx.x >> 5;
  bf16* pl = (bf16*)(smem + 2 * VCHUNK * 2) + wave * 512;  // per-wave P staging

  const int h = lane >> 4, n = lane & 15;
  const int wt = blockIdx.x * 2 + wave;        // query tile [0,512)
  const int bb = wt >> 7;                      // / (S/16); both waves same batch
  const int qbase = wt * 16;

  const bf16* qrow = Qb + (size_t)(qbase + n) * D;
  const bf16* vchunk0 = Vt + (size_t)(bb * KC) * VCHUNK;

  float m[8], l[8];
#pragma unroll
  for (int r = 0; r < 8; ++r) { m[r] = -1e30f; l[r] = 0.f; }

  v8f acc[UT];
#pragma unroll
  for (int ut = 0; ut < UT; ++ut)
#pragma unroll
    for (int r = 0; r < 8; ++r) acc[ut][r] = 0.f;

  // prologue: DMA chunk 0 into buffer 0
  if (wave == 0)
    tdm_load_vchunk(vchunk0, (unsigned int)(unsigned long long)vbuf);

  for (int kc = 0; kc < KC; ++kc) {
    const int cur = kc & 1;
    if (wave == 0)
      __builtin_amdgcn_s_wait_tensorcnt(0);    // chunk kc resident in vbuf[cur]
    __syncthreads();                           // publish to wave 1; prior reads done
    if (wave == 0 && kc + 1 < KC)
      tdm_load_vchunk(vchunk0 + (size_t)(kc + 1) * VCHUNK,
                      (unsigned int)(unsigned long long)(vbuf + (1 - cur) * VCHUNK));

    // ---- scores: s = (Q*scale*log2e) @ K^T, two 16x16 tiles (keys kc*32..+31)
    const bf16* kr0 = Kb + (size_t)(bb * S + kc * 32 + n) * U + h * 16;
    const bf16* kr1 = kr0 + (size_t)16 * U;
    v8f s0 = { 0.f, 0.f, 0.f, 0.f, 0.f, 0.f, 0.f, 0.f };
    v8f s1 = s0;
#pragma unroll
    for (int dt = 0; dt < DT; ++dt) {
      Frag a, b0, b1;
      a.h[0]  = *reinterpret_cast<const v8bf*>(qrow + dt * 32 + h * 16);
      a.h[1]  = *reinterpret_cast<const v8bf*>(qrow + dt * 32 + h * 16 + 8);
      b0.h[0] = *reinterpret_cast<const v8bf*>(kr0 + dt * 32);
      b0.h[1] = *reinterpret_cast<const v8bf*>(kr0 + dt * 32 + 8);
      b1.h[0] = *reinterpret_cast<const v8bf*>(kr1 + dt * 32);
      b1.h[1] = *reinterpret_cast<const v8bf*>(kr1 + dt * 32 + 8);
      s0 = wmma_bf16(a.f, b0.f, s0);
      s1 = wmma_bf16(a.f, b1.f, s1);
    }

    // ---- online softmax (base-2; scale folded into Q)
    float alpha[8], p0[8], p1[8];
#pragma unroll
    for (int r = 0; r < 8; ++r) {
      float v = fmaxf(s0[r], s1[r]);
      v = fmaxf(v, __shfl_xor(v, 1));
      v = fmaxf(v, __shfl_xor(v, 2));
      v = fmaxf(v, __shfl_xor(v, 4));
      v = fmaxf(v, __shfl_xor(v, 8));
      float mn = fmaxf(m[r], v);
      alpha[r] = __builtin_amdgcn_exp2f(m[r] - mn);
      m[r] = mn;
    }
#pragma unroll
    for (int r = 0; r < 8; ++r) {
      p0[r] = __builtin_amdgcn_exp2f(s0[r] - m[r]);
      p1[r] = __builtin_amdgcn_exp2f(s1[r] - m[r]);
      float sm = p0[r] + p1[r];
      sm += __shfl_xor(sm, 1);
      sm += __shfl_xor(sm, 2);
      sm += __shfl_xor(sm, 4);
      sm += __shfl_xor(sm, 8);
      l[r] = l[r] * alpha[r] + sm;
    }

    // ---- rescale running accumulators
#pragma unroll
    for (int ut = 0; ut < UT; ++ut)
#pragma unroll
      for (int r = 0; r < 8; ++r) acc[ut][r] *= alpha[r];

    // ---- P: C-layout -> A-fragment layout via wave-private LDS
    {
      const int i0 = n & 7;
      const int Lh = 8 * h + 16 * (n >> 3);
#pragma unroll
      for (int r = 0; r < 8; ++r) {
        int L = r + Lh;
        pl[L * 16 + i0]     = (bf16)p0[r];
        pl[L * 16 + 8 + i0] = (bf16)p1[r];
      }
    }
    asm volatile("s_wait_dscnt 0" ::: "memory");   // DS in-order; drain stores
    Frag pa;
    pa.h[0] = *reinterpret_cast<const v8bf*>(pl + lane * 16);
    pa.h[1] = *reinterpret_cast<const v8bf*>(pl + lane * 16 + 8);
    asm volatile("" ::: "memory");                 // keep loads before next-iter stores

    // ---- PV: acc += P(16x32) @ V(32x16) from the LDS-resident chunk
    const bf16* vfr = vbuf + cur * VCHUNK + lane * 16;
#pragma unroll
    for (int ut = 0; ut < UT; ++ut) {
      Frag vb;
      vb.h[0] = *reinterpret_cast<const v8bf*>(vfr + ut * 512);
      vb.h[1] = *reinterpret_cast<const v8bf*>(vfr + ut * 512 + 8);
      acc[ut] = wmma_bf16(pa.f, vb.f, acc[ut]);
    }
  }

  // ---- normalize and write f32 output
  float inv[8];
#pragma unroll
  for (int r = 0; r < 8; ++r) inv[r] = 1.0f / l[r];
#pragma unroll
  for (int ut = 0; ut < UT; ++ut)
#pragma unroll
    for (int r = 0; r < 8; ++r) {
      int row = qbase + r + 8 * h;
      out[(size_t)row * U + ut * 16 + n] = acc[ut][r] * inv[r];
    }
}

// ---------------------------------------------------------------------------
extern "C" void kernel_launch(void* const* d_in, const int* in_sizes, int n_in,
                              void* d_out, int out_size, void* d_ws, size_t ws_size,
                              hipStream_t stream) {
  const float* x  = (const float*)d_in[0];
  const float* Wq = (const float*)d_in[1];
  const float* Wk = (const float*)d_in[2];
  const float* Wv = (const float*)d_in[3];
  float* out = (float*)d_out;

  char* ws = (char*)d_ws;
  bf16* xb = (bf16*)(ws + XB_OFF);
  bf16* Wt = (bf16*)(ws + WT_OFF);
  bf16* Qb = (bf16*)(ws + QB_OFF);
  bf16* Kb = (bf16*)(ws + KB_OFF);
  bf16* Vt = (bf16*)(ws + VT_OFF);

  cvt_x_kernel<<<(R * D / 4 + 255) / 256, 256, 0, stream>>>(x, xb, R * D / 4);
  cvt_w_kernel<<<(3 * DT * UT * 32 + 127) / 128, 128, 0, stream>>>(Wq, Wk, Wv, Wt);
  qkv_gemm_kernel<<<dim3(R / 16, UT / 4, 3), 128, 0, stream>>>(xb, Wt, Qb, Kb, Vt);

  const size_t attn_lds = (size_t)2 * VCHUNK * 2 + 2 * 512 * 2;  // 100,352 bytes
  attn_kernel<<<R / 16 / 2, 64, attn_lds, stream>>>(Qb, Kb, Vt, out);
}
